// GraphConvolution_52158082842769
// MI455X (gfx1250) — matrix operand order repaired and checked by reading, not compiled, not run on hardware
//
#include <hip/hip_runtime.h>

#define N_ROWS 16384
#define FDIM   128
#define UNITS_ 128
#define KC     32
#define BLOCK_M 64
#define THREADS 128
#define XSTRIDE 40                      // padded per-column stride (elements) for X chunk in LDS (80 B, 16B-aligned)
#define WSTRIDE 136                     // padded stride for W / agg tiles in LDS
#define XBUF_ELEMS (FDIM * XSTRIDE)     // 5120 bf16 per buffer
#define LDSX_BYTES (2 * XBUF_ELEMS * 2) // 20480 (double buffered)
#define LDSW_ELEMS (UNITS_ * WSTRIDE)   // 17408
#define LDSW_BYTES (LDSW_ELEMS * 2)     // 34816
#define AGG_WAVE_ELEMS (16 * WSTRIDE)   // 2176
#define SMEM_BYTES (LDSX_BYTES + LDSW_BYTES) // 55296; agg region aliases X buffers
#define XT_BYTES ((size_t)FDIM * N_ROWS * 2) // 4 MB bf16 transposed X in d_ws

typedef __attribute__((ext_vector_type(16))) __bf16 v16bf;
typedef __attribute__((ext_vector_type(8)))  __bf16 v8bf;
typedef __attribute__((ext_vector_type(2)))  __bf16 v2bf;
typedef __attribute__((ext_vector_type(8)))  float  v8f;
typedef __attribute__((ext_vector_type(8)))  unsigned v8u;
typedef __attribute__((ext_vector_type(4)))  unsigned v4u;
typedef int v4i_vs __attribute__((vector_size(16)));   // matches builtin prototype

#if __has_builtin(__builtin_amdgcn_global_load_async_to_lds_b128)
#define HAVE_ASYNC_LDS 1
#else
#define HAVE_ASYNC_LDS 0
#endif

#define AS1 __attribute__((address_space(1)))
#define AS3 __attribute__((address_space(3)))

// ---- float pair -> packed 2x bf16: round-half-up (+0x8000) + one v_perm_b32 ----
static __device__ __forceinline__ unsigned packbf2(float lo, float hi) {
#if __has_builtin(__builtin_amdgcn_cvt_pk_bf16_f32)
  v2bf r = __builtin_amdgcn_cvt_pk_bf16_f32(lo, hi);
  return __builtin_bit_cast(unsigned, r);
#else
  unsigned a = __float_as_uint(lo) + 0x8000u;
  unsigned b = __float_as_uint(hi) + 0x8000u;
  return __builtin_amdgcn_perm(b, a, 0x07060302u);   // {b3,b2,a3,a2}
#endif
}
static __device__ __forceinline__ __bf16 tobf(float f) {   // cold paths only
  unsigned u = __float_as_uint(f) + 0x8000u;
  unsigned short h = (unsigned short)(u >> 16);
  __bf16 b; __builtin_memcpy(&b, &h, 2); return b;
}

#if HAVE_ASYNC_LDS
static __device__ __forceinline__ void async_cp16(const __bf16* g, __bf16* l) {
  AS1 v4i_vs* gp = (AS1 v4i_vs*)(unsigned long long)g;          // generic -> global (same value)
  AS3 v4i_vs* lp = (AS3 v4i_vs*)(unsigned)(unsigned long long)l; // generic -> local (low 32 bits)
  __builtin_amdgcn_global_load_async_to_lds_b128(gp, lp, 0, 0);
}
static __device__ __forceinline__ void wait_async0() {
#if __has_builtin(__builtin_amdgcn_s_wait_asynccnt)
  __builtin_amdgcn_s_wait_asynccnt(0);
#else
  asm volatile("s_wait_asynccnt 0" ::: "memory");
#endif
}
#endif

// ================= pre-pass: X[k][c] fp32 -> XT[c][k] bf16 (k-contiguous) ==========
__global__ __launch_bounds__(256)
void xt_prepass(const float* __restrict__ X, __bf16* __restrict__ XT) {
  int gid = blockIdx.x * 256 + threadIdx.x;       // 262144 threads
  int c  = gid & 127;
  int kb = (gid >> 7) << 3;                       // 8 consecutive k per thread
  float f[8];
#pragma unroll
  for (int i = 0; i < 8; ++i) f[i] = X[(size_t)(kb + i) * FDIM + c];  // coalesced reads
  v4u t;
#pragma unroll
  for (int i = 0; i < 4; ++i) t[i] = packbf2(f[2*i], f[2*i+1]);
  *(v4u*)&XT[(size_t)c * N_ROWS + kb] = t;        // 16B store per thread
}

// ================= main fused kernel (uses pre-transposed bf16 XT) =================
__global__ __launch_bounds__(THREADS)
void gcn_fused_ws(const __bf16* __restrict__ XT, const float* __restrict__ A,
                  const float* __restrict__ W, const float* __restrict__ bias,
                  float* __restrict__ out) {
  __shared__ __align__(128) unsigned char smem[SMEM_BYTES];
  __bf16* ldsX   = (__bf16*)smem;                    // GEMM1 X chunks (2 buffers)
  __bf16* ldsAgg = (__bf16*)smem;                    // reused after GEMM1
  __bf16* ldsW   = (__bf16*)(smem + LDSX_BYTES);

  const int tid   = threadIdx.x;
  const int wave  = tid >> 5;
  const int lane  = tid & 31;
  const int laneM = lane & 15;
  const int kGrp  = lane >> 4;
  const int rowBase = blockIdx.x * BLOCK_M + wave * 16;

  // ---- stage W into LDS, transposed [u][f], bf16 ----
  for (int i = tid; i < FDIM * UNITS_; i += THREADS) {
    int f = i >> 7, u = i & 127;
    ldsW[u * WSTRIDE + f] = tobf(W[i]);
  }

  v8f acc[8] = {};
  const float* aRow = A + (size_t)(rowBase + laneM) * (size_t)N_ROWS + (size_t)(kGrp * 8);
  const __bf16* xcol = XT + (size_t)tid * N_ROWS;     // column `tid`, k-contiguous bf16

  float4 ar[4];     // next A fragment, raw fp32

  // ---- prologue: chunk 0 ----
  {
    const float4* p0 = (const float4*)(aRow + 0);
    const float4* p1 = (const float4*)(aRow + 16);
    ar[0] = p0[0]; ar[1] = p0[1]; ar[2] = p1[0]; ar[3] = p1[1];
    __bf16* dst = ldsX + tid * XSTRIDE;
#if HAVE_ASYNC_LDS
#pragma unroll
    for (int q = 0; q < 4; ++q) async_cp16(xcol + 8*q, dst + 8*q);
    wait_async0();
#else
    v4u xst[4];
#pragma unroll
    for (int q = 0; q < 4; ++q) xst[q] = *(const v4u*)(xcol + 8*q);
#pragma unroll
    for (int q = 0; q < 4; ++q) *(v4u*)(dst + 8*q) = xst[q];
#endif
  }
  __syncthreads();

  const int NCHUNK = N_ROWS / KC;   // 512
  for (int ci = 0; ci < NCHUNK; ++ci) {
    const int cur = ci & 1;
    const bool hasNext = (ci + 1) < NCHUNK;

#if !HAVE_ASYNC_LDS
    v4u xst[4];
#endif
    if (hasNext) {
      const __bf16* src = xcol + (size_t)(ci + 1) * KC;
      __bf16* dst = ldsX + (cur ^ 1) * XBUF_ELEMS + tid * XSTRIDE;
#if HAVE_ASYNC_LDS
#pragma unroll
      for (int q = 0; q < 4; ++q) async_cp16(src + 8*q, dst + 8*q);   // overlaps with compute
#else
#pragma unroll
      for (int q = 0; q < 4; ++q) xst[q] = *(const v4u*)(src + 8*q);
#endif
    }

    // convert current A fragment: 8x (2 add + perm) -> v16bf
    v8u apk;
    apk[0] = packbf2(ar[0].x, ar[0].y);
    apk[1] = packbf2(ar[0].z, ar[0].w);
    apk[2] = packbf2(ar[1].x, ar[1].y);
    apk[3] = packbf2(ar[1].z, ar[1].w);
    apk[4] = packbf2(ar[2].x, ar[2].y);
    apk[5] = packbf2(ar[2].z, ar[2].w);
    apk[6] = packbf2(ar[3].x, ar[3].y);
    apk[7] = packbf2(ar[3].z, ar[3].w);
    v16bf afrag = __builtin_bit_cast(v16bf, apk);

    if (hasNext) {
      const float* ap = aRow + (size_t)(ci + 1) * KC;
      const float4* p0 = (const float4*)(ap + 0);
      const float4* p1 = (const float4*)(ap + 16);
      ar[0] = p0[0]; ar[1] = p0[1]; ar[2] = p1[0]; ar[3] = p1[1];
      if ((ci + 8) < NCHUNK)
        __builtin_prefetch(aRow + (size_t)(ci + 8) * KC, 0, 1);
    }

    // compute on current chunk (B fragments: 32x16 bf16 from transposed LDS)
    const __bf16* xb = ldsX + cur * XBUF_ELEMS;
#pragma unroll
    for (int nt = 0; nt < 8; ++nt) {
      const __bf16* p = xb + (nt * 16 + laneM) * XSTRIDE + kGrp * 8;
      v8bf lo = *(const v8bf*)p;
      v8bf hi = *(const v8bf*)(p + 16);
      v16bf bfrag = __builtin_shufflevector(lo, hi, 0,1,2,3,4,5,6,7,8,9,10,11,12,13,14,15);
      acc[nt] = __builtin_amdgcn_wmma_f32_16x16x32_bf16(false, afrag, false, bfrag,
                                                        (short)0, acc[nt], false, false);
    }

    if (hasNext) {
#if HAVE_ASYNC_LDS
      wait_async0();
#else
      __bf16* dst = ldsX + (cur ^ 1) * XBUF_ELEMS + tid * XSTRIDE;
#pragma unroll
      for (int q = 0; q < 4; ++q) *(v4u*)(dst + 8*q) = xst[q];
#endif
    }
    __syncthreads();   // single barrier per chunk (double-buffered)
  }

  // ---- move agg tile through per-wave LDS into A-fragment layout (bf16) ----
  __bf16* ag = ldsAgg + wave * AGG_WAVE_ELEMS;
#pragma unroll
  for (int nt = 0; nt < 8; ++nt) {
#pragma unroll
    for (int v = 0; v < 8; ++v) {
      ag[(v + 8 * kGrp) * WSTRIDE + nt * 16 + laneM] = tobf(acc[nt][v]);
    }
  }
  // per-wave LDS RAW is in-order; each wave reads only its own region

  // ---- GEMM2: out_tile = relu(agg_tile @ W + bias), K = 128 ----
  v8f acc2[8] = {};
#pragma unroll
  for (int kc2 = 0; kc2 < FDIM; kc2 += KC) {
    const __bf16* pa = ag + laneM * WSTRIDE + kc2 + kGrp * 8;
    v8bf alo = *(const v8bf*)pa;
    v8bf ahi = *(const v8bf*)(pa + 16);
    v16bf a2 = __builtin_shufflevector(alo, ahi, 0,1,2,3,4,5,6,7,8,9,10,11,12,13,14,15);
#pragma unroll
    for (int nt = 0; nt < 8; ++nt) {
      const __bf16* pb = ldsW + (nt * 16 + laneM) * WSTRIDE + kc2 + kGrp * 8;
      v8bf lo = *(const v8bf*)pb;
      v8bf hi = *(const v8bf*)(pb + 16);
      v16bf b2 = __builtin_shufflevector(lo, hi, 0,1,2,3,4,5,6,7,8,9,10,11,12,13,14,15);
      acc2[nt] = __builtin_amdgcn_wmma_f32_16x16x32_bf16(false, a2, false, b2,
                                                         (short)0, acc2[nt], false, false);
    }
  }

  // ---- epilogue: bias + relu + fp32 store ----
#pragma unroll
  for (int nt = 0; nt < 8; ++nt) {
    const int u = nt * 16 + laneM;
    const float bv = bias[u];
#pragma unroll
    for (int v = 0; v < 8; ++v) {
      const int m = rowBase + v + 8 * kGrp;
      float o = acc2[nt][v] + bv;
      o = fmaxf(o, 0.0f);
      out[(size_t)m * UNITS_ + u] = o;
    }
  }
}

// ================= fallback kernel (no workspace): round-3 self-contained ==========
__global__ __launch_bounds__(THREADS)
void gcn_fused_kernel(const float* __restrict__ X, const float* __restrict__ A,
                      const float* __restrict__ W, const float* __restrict__ bias,
                      float* __restrict__ out) {
  __shared__ __align__(128) unsigned char smem[SMEM_BYTES];
  __bf16* ldsX   = (__bf16*)smem;
  __bf16* ldsAgg = (__bf16*)smem;
  __bf16* ldsW   = (__bf16*)(smem + LDSX_BYTES);

  const int tid   = threadIdx.x;
  const int wave  = tid >> 5;
  const int lane  = tid & 31;
  const int laneM = lane & 15;
  const int kGrp  = lane >> 4;
  const int rowBase = blockIdx.x * BLOCK_M + wave * 16;

  for (int i = tid; i < FDIM * UNITS_; i += THREADS) {
    int f = i >> 7, u = i & 127;
    ldsW[u * WSTRIDE + f] = tobf(W[i]);
  }

  v8f acc[8] = {};
  const float* aRow = A + (size_t)(rowBase + laneM) * (size_t)N_ROWS + (size_t)(kGrp * 8);
  float4 ar[4];
  float  xr[32];

  {
    const float4* p0 = (const float4*)(aRow + 0);
    const float4* p1 = (const float4*)(aRow + 16);
    ar[0] = p0[0]; ar[1] = p0[1]; ar[2] = p1[0]; ar[3] = p1[1];
#pragma unroll
    for (int j = 0; j < 16; ++j) {
      xr[2*j]   = X[(size_t)(2*j)   * FDIM + tid];
      xr[2*j+1] = X[(size_t)(2*j+1) * FDIM + tid];
    }
    unsigned xpk[16];
#pragma unroll
    for (int j = 0; j < 16; ++j) xpk[j] = packbf2(xr[2*j], xr[2*j+1]);
#pragma unroll
    for (int q = 0; q < 4; ++q) {
      v4u t; t[0]=xpk[4*q]; t[1]=xpk[4*q+1]; t[2]=xpk[4*q+2]; t[3]=xpk[4*q+3];
      *(v4u*)&ldsX[tid * XSTRIDE + 8*q] = t;
    }
  }
  __syncthreads();

  const int NCHUNK = N_ROWS / KC;
  for (int ci = 0; ci < NCHUNK; ++ci) {
    const int cur = ci & 1;
    v8u apk;
    apk[0] = packbf2(ar[0].x, ar[0].y);
    apk[1] = packbf2(ar[0].z, ar[0].w);
    apk[2] = packbf2(ar[1].x, ar[1].y);
    apk[3] = packbf2(ar[1].z, ar[1].w);
    apk[4] = packbf2(ar[2].x, ar[2].y);
    apk[5] = packbf2(ar[2].z, ar[2].w);
    apk[6] = packbf2(ar[3].x, ar[3].y);
    apk[7] = packbf2(ar[3].z, ar[3].w);
    v16bf afrag = __builtin_bit_cast(v16bf, apk);

    const bool hasNext = (ci + 1) < NCHUNK;
    if (hasNext) {
      const float* ap = aRow + (size_t)(ci + 1) * KC;
      const float4* p0 = (const float4*)(ap + 0);
      const float4* p1 = (const float4*)(ap + 16);
      ar[0] = p0[0]; ar[1] = p0[1]; ar[2] = p1[0]; ar[3] = p1[1];
      const float* xp = X + (size_t)(ci + 1) * KC * FDIM + tid;
#pragma unroll
      for (int j = 0; j < 16; ++j) {
        xr[2*j]   = xp[(2*j)   * FDIM];
        xr[2*j+1] = xp[(2*j+1) * FDIM];
      }
      if ((ci + 8) < NCHUNK)
        __builtin_prefetch(aRow + (size_t)(ci + 8) * KC, 0, 1);
    }

    const __bf16* xb = ldsX + cur * XBUF_ELEMS;
#pragma unroll
    for (int nt = 0; nt < 8; ++nt) {
      const __bf16* p = xb + (nt * 16 + laneM) * XSTRIDE + kGrp * 8;
      v8bf lo = *(const v8bf*)p;
      v8bf hi = *(const v8bf*)(p + 16);
      v16bf bfrag = __builtin_shufflevector(lo, hi, 0,1,2,3,4,5,6,7,8,9,10,11,12,13,14,15);
      acc[nt] = __builtin_amdgcn_wmma_f32_16x16x32_bf16(false, afrag, false, bfrag,
                                                        (short)0, acc[nt], false, false);
    }

    if (hasNext) {
      __bf16* b = ldsX + (cur ^ 1) * XBUF_ELEMS;
      unsigned xpk[16];
#pragma unroll
      for (int j = 0; j < 16; ++j) xpk[j] = packbf2(xr[2*j], xr[2*j+1]);
#pragma unroll
      for (int q = 0; q < 4; ++q) {
        v4u t; t[0]=xpk[4*q]; t[1]=xpk[4*q+1]; t[2]=xpk[4*q+2]; t[3]=xpk[4*q+3];
        *(v4u*)&b[tid * XSTRIDE + 8*q] = t;
      }
    }
    __syncthreads();
  }

  __bf16* ag = ldsAgg + wave * AGG_WAVE_ELEMS;
#pragma unroll
  for (int nt = 0; nt < 8; ++nt)
#pragma unroll
    for (int v = 0; v < 8; ++v)
      ag[(v + 8 * kGrp) * WSTRIDE + nt * 16 + laneM] = tobf(acc[nt][v]);

  v8f acc2[8] = {};
#pragma unroll
  for (int kc2 = 0; kc2 < FDIM; kc2 += KC) {
    const __bf16* pa = ag + laneM * WSTRIDE + kc2 + kGrp * 8;
    v8bf alo = *(const v8bf*)pa;
    v8bf ahi = *(const v8bf*)(pa + 16);
    v16bf a2 = __builtin_shufflevector(alo, ahi, 0,1,2,3,4,5,6,7,8,9,10,11,12,13,14,15);
#pragma unroll
    for (int nt = 0; nt < 8; ++nt) {
      const __bf16* pb = ldsW + (nt * 16 + laneM) * WSTRIDE + kc2 + kGrp * 8;
      v8bf lo = *(const v8bf*)pb;
      v8bf hi = *(const v8bf*)(pb + 16);
      v16bf b2 = __builtin_shufflevector(lo, hi, 0,1,2,3,4,5,6,7,8,9,10,11,12,13,14,15);
      acc2[nt] = __builtin_amdgcn_wmma_f32_16x16x32_bf16(false, a2, false, b2,
                                                         (short)0, acc2[nt], false, false);
    }
  }

#pragma unroll
  for (int nt = 0; nt < 8; ++nt) {
    const int u = nt * 16 + laneM;
    const float bv = bias[u];
#pragma unroll
    for (int v = 0; v < 8; ++v) {
      const int m = rowBase + v + 8 * kGrp;
      float o = acc2[nt][v] + bv;
      o = fmaxf(o, 0.0f);
      out[(size_t)m * UNITS_ + u] = o;
    }
  }
}

extern "C" void kernel_launch(void* const* d_in, const int* in_sizes, int n_in,
                              void* d_out, int out_size, void* d_ws, size_t ws_size,
                              hipStream_t stream) {
  (void)in_sizes; (void)n_in; (void)out_size;
  const float* X    = (const float*)d_in[0];   // features [N, F]
  const float* A    = (const float*)d_in[1];   // adjacency [N, N]
  const float* W    = (const float*)d_in[2];   // kernel [F, UNITS]
  const float* bias = (const float*)d_in[3];   // [UNITS]
  float* out = (float*)d_out;                  // [N, UNITS] fp32

  dim3 grid(N_ROWS / BLOCK_M);   // 256 workgroups
  dim3 block(THREADS);           // 128 threads = 4 wave32

  if (ws_size >= XT_BYTES) {
    __bf16* XT = (__bf16*)d_ws;
    hipLaunchKernelGGL(xt_prepass, dim3((N_ROWS / 8) * FDIM / 256), dim3(256), 0, stream, X, XT);
    hipLaunchKernelGGL(gcn_fused_ws, grid, block, 0, stream, XT, A, W, bias, out);
  } else {
    hipLaunchKernelGGL(gcn_fused_kernel, grid, block, 0, stream, X, A, W, bias, out);
  }
}
